// PxlSimCNN_47425028882919
// MI455X (gfx1250) — matrix operand order repaired and checked by reading, not compiled, not run on hardware
//
#include <hip/hip_runtime.h>

// ---------------------------------------------------------------------------
// MI455X / gfx1250 implementation.
// Pipeline:
//   1. k_pad_input : x fp32 -> zero-padded bf16 (halo 27 = max pad), 118x118
//   2. k_pack_w    : per-conv weights fp32 -> bf16 [oc][Kpad] rows (Kpad %32==0)
//   3. k_conv_wmma : implicit-GEMM conv, v_wmma_f32_16x16x32_bf16,
//                    4 oc-tiles register-blocked per wave, B im2col in LDS,
//                    global_prefetch of next A fragment
//   4. k_gram_pool : fused gram + 5x5/stride2 avg-pool: G = (1/25) F F^T,
//                    F = 32ch x 25window via 2x2 tiles of 16x16x32 WMMA
//   5. k_cos_sim   : neighbor cosine sim over 1024-dim pooled grams,
//                    wave32 shuffle reductions, group mean
// ---------------------------------------------------------------------------

typedef __attribute__((ext_vector_type(16))) __bf16        v16bf;
typedef __attribute__((ext_vector_type(8)))  float         v8f;
typedef __attribute__((ext_vector_type(4)))  unsigned int  v4u;

union Frag {
  v16bf          bf;
  v4u            q[2];
  unsigned short h[16];
};

__device__ __constant__ int c_fs[8]   = {3, 5, 7, 11, 15, 23, 37, 55};
__device__ __constant__ int c_K[8]    = {27, 75, 147, 363, 675, 1587, 4107, 9075};
__device__ __constant__ int c_Kpad[8] = {32, 96, 160, 384, 704, 1600, 4128, 9088};
__device__ __constant__ int c_woff[8] = {0, 2048, 8192, 18432, 43008, 88064, 190464, 454656};

static __device__ __forceinline__ unsigned short f2bf(float f) {
  unsigned int u = __float_as_uint(f);
  u += 0x7FFFu + ((u >> 16) & 1u);   // round-to-nearest-even
  return (unsigned short)(u >> 16);
}
static __device__ __forceinline__ float bf2f(unsigned short h) {
  return __uint_as_float(((unsigned int)h) << 16);
}

// ---------------- 1. pad + convert input -----------------------------------
// xpad: (2,3,118,118) bf16, halo of 27 (max (55-1)/2) on each side.
__global__ void k_pad_input(const float* __restrict__ x,
                            unsigned short* __restrict__ xpad) {
  int idx = blockIdx.x * 256 + threadIdx.x;
  const int NP = 2 * 3 * 118 * 118;
  if (idx >= NP) return;
  int xx = idx % 118; int t = idx / 118;
  int yy = t % 118;   t /= 118;
  int c  = t % 3;     int b = t / 3;
  float v = 0.f;
  if (yy >= 27 && yy < 91 && xx >= 27 && xx < 91)
    v = x[((b * 3 + c) * 64 + (yy - 27)) * 64 + (xx - 27)];
  xpad[idx] = f2bf(v);
}

// ---------------- 2. pack weights bf16 [oc][Kpad] --------------------------
struct WPtrs { const float* w[8]; };
struct BPtrs { const float* b[8]; };

__global__ void k_pack_w(WPtrs wp, unsigned short* __restrict__ wpack) {
  int e = blockIdx.x * 256 + threadIdx.x;
  if (e >= 1036288) return;
  int i = 7;
  for (int j = 0; j < 7; ++j) { if (e < c_woff[j + 1]) { i = j; break; } }
  int rel = e - c_woff[i];
  int Kp = c_Kpad[i], K = c_K[i];
  int oc = rel / Kp, k = rel - oc * Kp;
  float v = (k < K) ? wp.w[i][oc * K + k] : 0.f;   // w layout [oc][ic][f][f] flat == oc*K + k
  wpack[e] = f2bf(v);
}

// ---------------- 3. implicit-GEMM conv via WMMA ---------------------------
// grid = (32 pixel-tiles of 128, 2 batches, 8 convs), block = 256 (8 waves).
// wave w owns pixel sub-tile w (16 pixels) and register-blocks all 4 oc-tiles:
// per K-step: 1 B-frag from LDS, 4 A-frags from packed weights, 4 WMMAs.
__global__ __launch_bounds__(256)
void k_conv_wmma(const unsigned short* __restrict__ xpad,
                 const unsigned short* __restrict__ wpack,
                 BPtrs bp,
                 unsigned short* __restrict__ feats) {
  __shared__ unsigned short ldsB[128 * 32];  // [pix 0..127][k 0..31], 8 KB
  const int tid  = threadIdx.x;
  const int lane = tid & 31;
  const int wv   = tid >> 5;
  const int conv = blockIdx.z;
  const int b    = blockIdx.y;
  const int p0   = blockIdx.x * 128;
  const int f    = c_fs[conv];
  const int K    = c_K[conv];
  const int Kp   = c_Kpad[conv];
  const int pad  = (f - 1) >> 1;
  const int ff   = f * f;
  const int bbase = b * (3 * 118 * 118);

  const int m  = lane & 15;
  const int hi = lane >> 4;

  // A fragment bases for the 4 oc-tiles (16-bit A layout:
  // lane(m,hi) holds K = hi*8+0..7 and 16+hi*8+0..7).
  const unsigned short* Abase = wpack + c_woff[conv] + m * Kp + hi * 8;

  // staging: this thread's fixed k-lane and pixel stripe
  const int kl      = tid & 31;   // fixed k within the 32-wide K-step
  const int pixbase = tid >> 5;   // pixels pixbase + 8*i, i = 0..15

  v8f acc0 = {}, acc1 = {}, acc2 = {}, acc3 = {};
  const int nkb = Kp >> 5;
  for (int kb = 0; kb < nkb; ++kb) {
    // ---- stage im2col B tile (32 k x 128 pixels) into LDS ----
    // decompose k once per K-step per thread; per-pixel address is linear.
    const int k = kb * 32 + kl;
    int koff = 0, kvalid = 0;
    if (k < K) {
      int ic = k / ff, r = k - ic * ff;
      int ky = r / f,  kx = r - ky * f;
      koff = (ic * 118 + (ky - pad + 27)) * 118 + (kx - pad + 27);
      kvalid = 1;
    }
    #pragma unroll
    for (int i = 0; i < 16; ++i) {
      int pix = pixbase + i * 8;
      int p   = p0 + pix;
      int y   = p >> 6, xq = p & 63;
      unsigned short v = 0;
      if (kvalid) v = xpad[bbase + koff + y * 118 + xq];
      ldsB[pix * 32 + kl] = v;
    }
    __syncthreads();

    // prefetch next K-step's A fragments while this step's WMMAs run
    if (kb + 1 < nkb) {
      __builtin_prefetch(Abase + (kb + 1) * 32, 0, 0);
      __builtin_prefetch(Abase + 48 * Kp + (kb + 1) * 32, 0, 0);
    }

    // B fragment (16-bit B layout): lane(n,hi) holds K = hi*8+0..7 and 16+hi*8+0..7
    Frag bb;
    const unsigned short* bl = &ldsB[(wv * 16 + m) * 32 + hi * 8];
    bb.q[0] = *(const v4u*)(bl);
    bb.q[1] = *(const v4u*)(bl + 16);

    Frag a;
    const unsigned short* ap = Abase + kb * 32;
    a.q[0] = *(const v4u*)(ap);
    a.q[1] = *(const v4u*)(ap + 16);
    acc0 = __builtin_amdgcn_wmma_f32_16x16x32_bf16(false, a.bf, false, bb.bf, (short)0, acc0, false, false);
    ap += 16 * Kp;
    a.q[0] = *(const v4u*)(ap);
    a.q[1] = *(const v4u*)(ap + 16);
    acc1 = __builtin_amdgcn_wmma_f32_16x16x32_bf16(false, a.bf, false, bb.bf, (short)0, acc1, false, false);
    ap += 16 * Kp;
    a.q[0] = *(const v4u*)(ap);
    a.q[1] = *(const v4u*)(ap + 16);
    acc2 = __builtin_amdgcn_wmma_f32_16x16x32_bf16(false, a.bf, false, bb.bf, (short)0, acc2, false, false);
    ap += 16 * Kp;
    a.q[0] = *(const v4u*)(ap);
    a.q[1] = *(const v4u*)(ap + 16);
    acc3 = __builtin_amdgcn_wmma_f32_16x16x32_bf16(false, a.bf, false, bb.bf, (short)0, acc3, false, false);
    __syncthreads();
  }

  // epilogue: bias + ReLU, store bf16 feats [b][512][4096]
  const int pixn = p0 + wv * 16 + (lane & 15);
  const float* bias = bp.b[conv];
  v8f accs[4] = {acc0, acc1, acc2, acc3};
  #pragma unroll
  for (int t = 0; t < 4; ++t) {
    #pragma unroll
    for (int r = 0; r < 8; ++r) {
      int ocl = t * 16 + r + hi * 8;              // C/D layout: vgpr r, lane-half hi
      float v = accs[t][r] + bias[ocl];
      v = v > 0.f ? v : 0.f;
      feats[((size_t)b * 512 + conv * 64 + ocl) * 4096 + pixn] = f2bf(v);
    }
  }
}

// ---------------- 4. fused gram + 5x5 avg-pool via WMMA --------------------
// pooled G(p) = (1/25) F F^T,  F = 32ch x 25 window pixels (K padded to 32).
// grid = (32 y', 32 slices), block = 256 (8 waves, 4 x' each).
__global__ __launch_bounds__(256)
void k_gram_pool(const unsigned short* __restrict__ feats,
                 unsigned short* __restrict__ gram) {
  __shared__ unsigned short patch[32 * 5 * 68];  // [ch][dy 0..4][col 0..67], 21.76 KB
  const int tid  = threadIdx.x;
  const int lane = tid & 31;
  const int wv   = tid >> 5;
  const int yP   = blockIdx.x;       // y' in 0..31
  const int n    = blockIdx.y;       // slice = b*16 + group
  const int b    = n >> 4;
  const int chb  = (n & 15) * 32;
  const int y0   = yP * 2;

  for (int idx = tid; idx < 32 * 5 * 68; idx += 256) {
    int ch  = idx / 340;
    int r   = idx - ch * 340;
    int dy  = r / 68;
    int col = r - dy * 68;
    int yin = y0 - 2 + dy;
    int xin = col - 2;
    unsigned short v = 0;
    if (yin >= 0 && yin < 64 && xin >= 0 && xin < 64)
      v = feats[((size_t)b * 512 + chb + ch) * 4096 + yin * 64 + xin];
    patch[idx] = v;
  }
  __syncthreads();

  const int m  = lane & 15;
  const int hi = lane >> 4;
  for (int xi = 0; xi < 4; ++xi) {
    const int xP = wv * 4 + xi;      // x' in 0..31
    Frag A[2], B[2];
    #pragma unroll
    for (int s = 0; s < 2; ++s) {
      const int chO = (s * 16 + m) * 340;
      #pragma unroll
      for (int j = 0; j < 16; ++j) {
        int kA = (j < 8) ? (hi * 8 + j) : (16 + hi * 8 + (j - 8));
        unsigned short va = 0;
        if (kA < 25) { int dy = kA / 5, dx = kA - dy * 5; va = patch[chO + dy * 68 + xP * 2 + dx]; }
        A[s].h[j] = va;
        int kB = (j < 8) ? (hi * 8 + j) : (16 + hi * 8 + (j - 8));
        unsigned short vb = 0;
        if (kB < 25) { int dy = kB / 5, dx = kB - dy * 5; vb = patch[chO + dy * 68 + xP * 2 + dx]; }
        B[s].h[j] = vb;
      }
    }
    v8f c00 = {}, c01 = {}, c10 = {}, c11 = {};
    c00 = __builtin_amdgcn_wmma_f32_16x16x32_bf16(false, A[0].bf, false, B[0].bf, (short)0, c00, false, false);
    c01 = __builtin_amdgcn_wmma_f32_16x16x32_bf16(false, A[0].bf, false, B[1].bf, (short)0, c01, false, false);
    c10 = __builtin_amdgcn_wmma_f32_16x16x32_bf16(false, A[1].bf, false, B[0].bf, (short)0, c10, false, false);
    c11 = __builtin_amdgcn_wmma_f32_16x16x32_bf16(false, A[1].bf, false, B[1].bf, (short)0, c11, false, false);

    unsigned short* gp = gram + (((size_t)n * 32 + yP) * 32 + xP) * 1024;
    const int j0 = lane & 15;
    #pragma unroll
    for (int r = 0; r < 8; ++r) {
      int i0 = r + hi * 8;
      gp[i0 * 32 + j0]             = f2bf(c00[r] * 0.04f);
      gp[i0 * 32 + 16 + j0]        = f2bf(c01[r] * 0.04f);
      gp[(16 + i0) * 32 + j0]      = f2bf(c10[r] * 0.04f);
      gp[(16 + i0) * 32 + 16 + j0] = f2bf(c11[r] * 0.04f);
    }
  }
}

// ---------------- 5. neighbor cosine similarity + group mean ---------------
// one wave32 per output pixel; grid = 256 blocks x 8 waves = 2048 pixels.
__global__ __launch_bounds__(256)
void k_cos_sim(const unsigned short* __restrict__ gram,
               float* __restrict__ out) {
  const int lane = threadIdx.x & 31;
  const int wv   = threadIdx.x >> 5;
  const int wid  = blockIdx.x * 8 + wv;   // 0..2047
  const int b    = wid >> 10;
  const int rem  = wid & 1023;
  const int y    = rem >> 5;
  const int x    = rem & 31;
  const int xl = (x + 1 < 32) ? x + 1 : 31;   // left-shift neighbor (replicate)
  const int xr = (x - 1 >= 0) ? x - 1 : 0;    // right-shift neighbor
  const int yt = (y + 1 < 32) ? y + 1 : 31;
  const int yb = (y - 1 >= 0) ? y - 1 : 0;

  float acc = 0.f;
  for (int g = 0; g < 16; ++g) {
    const int n = b * 16 + g;
    const unsigned short* L = gram + (((size_t)n * 32 + y ) * 32 + xl) * 1024;
    const unsigned short* R = gram + (((size_t)n * 32 + y ) * 32 + xr) * 1024;
    const unsigned short* T = gram + (((size_t)n * 32 + yt) * 32 + x ) * 1024;
    const unsigned short* D = gram + (((size_t)n * 32 + yb) * 32 + x ) * 1024;
    float hn = 0, l2 = 0, r2 = 0, vn = 0, t2 = 0, d2 = 0;
    for (int i = lane; i < 1024; i += 32) {
      float lv = bf2f(L[i]), rv = bf2f(R[i]);
      float tv = bf2f(T[i]), dv = bf2f(D[i]);
      hn += lv * rv; l2 += lv * lv; r2 += rv * rv;
      vn += tv * dv; t2 += tv * tv; d2 += dv * dv;
    }
    #pragma unroll
    for (int o = 16; o > 0; o >>= 1) {
      hn += __shfl_xor(hn, o, 32);
      l2 += __shfl_xor(l2, o, 32);
      r2 += __shfl_xor(r2, o, 32);
      vn += __shfl_xor(vn, o, 32);
      t2 += __shfl_xor(t2, o, 32);
      d2 += __shfl_xor(d2, o, 32);
    }
    float ch = hn / (sqrtf(l2) * sqrtf(r2) + 1e-12f);
    float cv = vn / (sqrtf(t2) * sqrtf(d2) + 1e-12f);
    acc += 0.5f * (ch + cv);
  }
  if (lane == 0) out[b * 1024 + y * 32 + x] = acc * 0.0625f;
}

// ---------------- launch ----------------------------------------------------
extern "C" void kernel_launch(void* const* d_in, const int* in_sizes, int n_in,
                              void* d_out, int out_size, void* d_ws, size_t ws_size,
                              hipStream_t stream) {
  (void)in_sizes; (void)n_in; (void)out_size; (void)ws_size;
  const float* x = (const float*)d_in[0];
  WPtrs wp; BPtrs bp;
  for (int i = 0; i < 8; ++i) {
    wp.w[i] = (const float*)d_in[1 + 2 * i];
    bp.b[i] = (const float*)d_in[2 + 2 * i];
  }

  // workspace layout (bytes), all 256B aligned:
  //   [0)          xpad   bf16  2*3*118*118      = 167,088 B
  //   [167168)     wpack  bf16  1,036,288 elems  = 2,072,576 B
  //   [2239744)    feats  bf16  2*512*64*64      = 8,388,608 B
  //   [10628352)   gram   bf16  32*32*32*1024    = 67,108,864 B  (end ~77.7 MB)
  char* ws = (char*)d_ws;
  unsigned short* xpad  = (unsigned short*)(ws + 0);
  unsigned short* wpack = (unsigned short*)(ws + 167168);
  unsigned short* feats = (unsigned short*)(ws + 2239744);
  unsigned short* gram  = (unsigned short*)(ws + 10628352);
  float* out = (float*)d_out;

  k_pad_input<<<dim3((2 * 3 * 118 * 118 + 255) / 256), dim3(256), 0, stream>>>(x, xpad);
  k_pack_w  <<<dim3(1036288 / 256),                    dim3(256), 0, stream>>>(wp, wpack);
  k_conv_wmma<<<dim3(32, 2, 8),                        dim3(256), 0, stream>>>(xpad, wpack, bp, feats);
  k_gram_pool<<<dim3(32, 32),                          dim3(256), 0, stream>>>(feats, gram);
  k_cos_sim <<<dim3(256),                              dim3(256), 0, stream>>>(gram, out);
}
